// Sen_Att_23639499997694
// MI455X (gfx1250) — matrix operand order repaired and checked by reading, not compiled, not run on hardware
//
#include <hip/hip_runtime.h>
#include <hip/hip_bf16.h>

// Problem dims (fixed by the reference)
#define D_IN  1024
#define D_K   2048
#define BATCH 8
#define NSEQ  2048
#define MTOT  (BATCH * NSEQ)   // 16384 rows total
#define KSTEP 32

typedef __attribute__((ext_vector_type(16))) __bf16 v16bf;
typedef __attribute__((ext_vector_type(8)))  __bf16 v8bf;
typedef __attribute__((ext_vector_type(8)))  float  v8f;

typedef __attribute__((ext_vector_type(4))) unsigned int u32x4;
typedef __attribute__((ext_vector_type(8))) int i32x8;
typedef __attribute__((ext_vector_type(4))) int i32x4;

#if __has_builtin(__builtin_amdgcn_tensor_load_to_lds) && \
    __has_builtin(__builtin_amdgcn_s_wait_tensorcnt)
#define HAVE_TDM 1
#else
#define HAVE_TDM 0
#endif

#if HAVE_TDM
// ---------------------------------------------------------------------------
// Issue one 2D TDM load: tile_d1 rows x tile_d0 elements (bf16, data_size=2B),
// row stride stride0 (elements) in global memory, packed row-major into LDS
// at byte offset lds_addr. D# bitfields per CDNA5 ISA ch.8 (Group0/Group1).
// ---------------------------------------------------------------------------
__device__ __forceinline__ void tdm_load_2d(unsigned lds_addr, const void* gptr,
                                            unsigned tile_d0, unsigned tile_d1,
                                            unsigned stride0) {
  unsigned long long ga = (unsigned long long)(uintptr_t)gptr;
  u32x4 g0;
  g0[0] = 1u;                                            // count=1, user mode
  g0[1] = lds_addr;                                      // lds_addr[31:0]
  g0[2] = (unsigned)(ga & 0xffffffffu);                  // global_addr[31:0]
  g0[3] = (unsigned)((ga >> 32) & 0x01ffffffu)           // global_addr[56:32]
          | (2u << 30);                                  // type=2 ("image")
  i32x8 g1;
  g1[0] = (int)(1u << 16);                               // wg_mask=0, data_size=1 (2B)
  g1[1] = (int)((tile_d0 & 0xffffu) << 16);              // tensor_dim0[15:0]
  g1[2] = (int)((tile_d0 >> 16) |                        // tensor_dim0[31:16]
                ((tile_d1 & 0xffffu) << 16));            // tensor_dim1[15:0]
  g1[3] = (int)((tile_d1 >> 16) |                        // tensor_dim1[31:16]
                ((tile_d0 & 0xffffu) << 16));            // tile_dim0
  g1[4] = (int)(tile_d1 & 0xffffu);                      // tile_dim1; tile_dim2=0
  g1[5] = (int)stride0;                                  // tensor_dim0_stride[31:0]
  g1[6] = 0;                                             // stride0 hi / stride1 lo
  g1[7] = 0;
  i32x4 z = {0, 0, 0, 0};
#if __clang_major__ >= 23
  i32x8 z8 = {0, 0, 0, 0, 0, 0, 0, 0};
  __builtin_amdgcn_tensor_load_to_lds(g0, g1, z, z, z8, 0);
#else
  __builtin_amdgcn_tensor_load_to_lds(g0, g1, z, z, 0);
#endif
}
#endif  // HAVE_TDM

// ---------------------------------------------------------------------------
// Elementwise f32 -> bf16 cast
// ---------------------------------------------------------------------------
__global__ __launch_bounds__(256)
void k_cast_bf16(const float* __restrict__ in, __bf16* __restrict__ out, int n) {
  int i = blockIdx.x * blockDim.x + threadIdx.x;
  if (i < n) out[i] = (__bf16)in[i];
}

// ---------------------------------------------------------------------------
// Transpose f32 [R][C] -> bf16 [C][R] via LDS tile (pad to dodge bank conflicts)
// ---------------------------------------------------------------------------
__global__ __launch_bounds__(256)
void k_transpose_bf16(const float* __restrict__ W, __bf16* __restrict__ Wt,
                      int R, int C) {
  __shared__ float tile[32][33];
  const int tx = threadIdx.x & 31;
  const int ty = threadIdx.x >> 5;         // 0..7
  const int c0 = blockIdx.x * 32;
  const int r0 = blockIdx.y * 32;
#pragma unroll
  for (int i = 0; i < 32; i += 8) {
    int r = r0 + ty + i, c = c0 + tx;
    tile[ty + i][tx] = (r < R && c < C) ? W[(size_t)r * C + c] : 0.f;
  }
  __syncthreads();
#pragma unroll
  for (int i = 0; i < 32; i += 8) {
    int c = c0 + ty + i, r = r0 + tx;
    if (r < R && c < C) Wt[(size_t)c * R + r] = (__bf16)tile[tx][ty + i];
  }
}

// ---------------------------------------------------------------------------
// Generic bf16 WMMA GEMM: C = A(MxK, row-major, lda) * B(KxN, col-major: B
// element [k][n] stored at Bc[n*ldb + k]).
// Block = 128 threads = 4 waves; block tile 64(M) x 64(N), K stepped by 32.
// TDM double-buffered LDS pipeline: wave 0 issues tensor_load_to_lds for the
// next A/B slabs (tracked by TENSORcnt) while all waves run
// v_wmma_f32_16x16x32_bf16 on the current slab from LDS (ds_load_b128).
// mode: 0 = bf16 row-major out, 1 = bf16 transposed out (C[col*ldc+row]),
//       2 = f32 row-major out.
// Grid: (M/64, N/64).
// ---------------------------------------------------------------------------
__global__ __launch_bounds__(128)
void k_gemm_wmma(const __bf16* __restrict__ A, int lda,
                 const __bf16* __restrict__ B, int ldb,
                 void* __restrict__ Cv, int ldc,
                 int K, int mode) {
  __shared__ alignas(64) __bf16 sA[2][64 * KSTEP];   // 4 KiB x2
  __shared__ alignas(64) __bf16 sB[2][64 * KSTEP];   // 4 KiB x2

  const int lane = threadIdx.x & 31;
  const int wave = threadIdx.x >> 5;            // 0..3
  const int row0b = blockIdx.x * 64;
  const int col0  = blockIdx.y * 64;
  const int hi   = lane >> 4;                   // 0: lanes 0-15, 1: 16-31
  const int koff_a = hi << 4;                   // 0 / 16
  const int koff_b = hi << 4;                   // 0 / 16
  const int bcol = lane & 15;
  const int arow = lane & 15;

  const __bf16* gA = A + (size_t)row0b * lda;   // A slab origin (64 rows)
  const __bf16* gB = B + (size_t)col0 * ldb;    // B slab origin (64 columns)

  auto stage = [&](int buf, int k0) {
#if HAVE_TDM
    if (wave == 0) {
      tdm_load_2d((unsigned)(uintptr_t)(void*)&sA[buf][0], gA + k0,
                  KSTEP, 64, (unsigned)lda);
      tdm_load_2d((unsigned)(uintptr_t)(void*)&sB[buf][0], gB + k0,
                  KSTEP, 64, (unsigned)ldb);
    }
#else
    // Fallback: cooperative 16B-chunk staging (256 chunks over 128 threads).
    for (int c = threadIdx.x; c < 256; c += 128) {
      int row = c >> 2, part = (c & 3) << 3;
      *(v8bf*)&sA[buf][row * KSTEP + part] =
          *(const v8bf*)(gA + (size_t)row * lda + k0 + part);
      *(v8bf*)&sB[buf][row * KSTEP + part] =
          *(const v8bf*)(gB + (size_t)row * ldb + k0 + part);
    }
#endif
  };

  stage(0, 0);

  v8f acc[4] = {};
  const int nk = K / KSTEP;
  for (int i = 0; i < nk; ++i) {
    const int buf = i & 1;
#if HAVE_TDM
    if (wave == 0) __builtin_amdgcn_s_wait_tensorcnt(0);  // slab landed in LDS
#endif
    __syncthreads();
    if (i + 1 < nk) stage(buf ^ 1, (i + 1) * KSTEP);      // prefetch next slab

    // A fragment: this wave's 16-row window of the 64-row slab.
    const __bf16* sa = &sA[buf][(wave * 16 + arow) * KSTEP];
    v8bf alo = *(const v8bf*)(sa + koff_a);
    v8bf ahi = *(const v8bf*)(sa + koff_a + 8);
    v16bf afrag;
#pragma unroll
    for (int e = 0; e < 8; ++e) { afrag[e] = alo[e]; afrag[e + 8] = ahi[e]; }

#pragma unroll
    for (int t = 0; t < 4; ++t) {
      v16bf bfrag = *(const v16bf*)(&sB[buf][(t * 16 + bcol) * KSTEP + koff_b]);
      acc[t] = __builtin_amdgcn_wmma_f32_16x16x32_bf16(
          false, afrag, false, bfrag, (short)0, acc[t], false, false);
    }
    __syncthreads();   // all reads of buf done before it is restaged
  }

  // D layout: lane holds column (lane&15); vgpr r holds row r + 8*hi.
  const int row0 = row0b + wave * 16;
  const int crow = row0 + (hi << 3);
  if (mode == 0) {
    __bf16* C = (__bf16*)Cv;
#pragma unroll
    for (int t = 0; t < 4; ++t) {
      int col = col0 + t * 16 + bcol;
#pragma unroll
      for (int r = 0; r < 8; ++r)
        C[(size_t)(crow + r) * ldc + col] = (__bf16)acc[t][r];
    }
  } else if (mode == 1) {
    __bf16* C = (__bf16*)Cv;
#pragma unroll
    for (int t = 0; t < 4; ++t) {
      int col = col0 + t * 16 + bcol;
      v8bf pk;
#pragma unroll
      for (int r = 0; r < 8; ++r) pk[r] = (__bf16)acc[t][r];
      *(v8bf*)(C + (size_t)col * ldc + crow) = pk;   // 8 consecutive rows
    }
  } else {
    float* C = (float*)Cv;
#pragma unroll
    for (int t = 0; t < 4; ++t) {
      int col = col0 + t * 16 + bcol;
#pragma unroll
      for (int r = 0; r < 8; ++r)
        C[(size_t)(crow + r) * ldc + col] = acc[t][r];
    }
  }
}

// ---------------------------------------------------------------------------
// Row softmax over one 2048-wide row per block; f32 in, bf16 out.
// ---------------------------------------------------------------------------
__global__ __launch_bounds__(256)
void k_softmax_row(const float* __restrict__ S, __bf16* __restrict__ P) {
  const int row = blockIdx.x;
  const float* s = S + (size_t)row * NSEQ;
  __bf16* p = P + (size_t)row * NSEQ;
  const int tid = threadIdx.x;

  float v[8];
  float m = -3.402823466e38f;
#pragma unroll
  for (int i = 0; i < 8; ++i) { v[i] = s[tid + (i << 8)]; m = fmaxf(m, v[i]); }
#pragma unroll
  for (int off = 16; off > 0; off >>= 1) m = fmaxf(m, __shfl_xor(m, off, 32));

  __shared__ float smax[8];
  __shared__ float ssum[8];
  if ((tid & 31) == 0) smax[tid >> 5] = m;
  __syncthreads();
  m = smax[0];
#pragma unroll
  for (int w = 1; w < 8; ++w) m = fmaxf(m, smax[w]);

  float sum = 0.f;
#pragma unroll
  for (int i = 0; i < 8; ++i) { v[i] = __expf(v[i] - m); sum += v[i]; }
#pragma unroll
  for (int off = 16; off > 0; off >>= 1) sum += __shfl_xor(sum, off, 32);
  if ((tid & 31) == 0) ssum[tid >> 5] = sum;
  __syncthreads();
  sum = 0.f;
#pragma unroll
  for (int w = 0; w < 8; ++w) sum += ssum[w];

  float inv = 1.0f / sum;
#pragma unroll
  for (int i = 0; i < 8; ++i) p[tid + (i << 8)] = (__bf16)(v[i] * inv);
}

// ---------------------------------------------------------------------------
// Fused FC (+bias, ReLU) + residual + LayerNorm.
// Block = 256 threads (8 waves) computes 16 rows x 1024 cols. Wave w owns
// columns [w*128, w*128+128) as 8 WMMA accumulators; K = 2048 (64 steps).
// LN stats via half-wave shuffle reduction + LDS ds_add_f32 atomics.
// ---------------------------------------------------------------------------
__global__ __launch_bounds__(256)
void k_fc_ln(const __bf16* __restrict__ Ax,    // att_x [MTOT][D_K] bf16
             const __bf16* __restrict__ Wt,    // WfcT  [D_IN][D_K] bf16
             const float*  __restrict__ bfc,   // [D_IN]
             const float*  __restrict__ x,     // [MTOT][D_IN] f32
             const float*  __restrict__ gamma,
             const float*  __restrict__ beta,
             float* __restrict__ out) {        // [MTOT][D_IN] f32
  const int lane = threadIdx.x & 31;
  const int wave = threadIdx.x >> 5;           // 0..7
  const int row0 = blockIdx.x << 4;            // 16 rows per block
  const int hi = lane >> 4;
  const int koff_a = hi << 4;
  const int koff_b = hi << 4;
  const int bcol = lane & 15;
  const int colw = wave << 7;                  // wave * 128
  const __bf16* pa = Ax + (size_t)(row0 + (lane & 15)) * D_K;

  v8f acc[8] = {};
  for (int k0 = 0; k0 < D_K; k0 += KSTEP) {
    __builtin_prefetch(pa + k0 + 128, 0, 3);
    v8bf alo = *(const v8bf*)(pa + k0 + koff_a);
    v8bf ahi = *(const v8bf*)(pa + k0 + koff_a + 8);
    v16bf a;
#pragma unroll
    for (int e = 0; e < 8; ++e) { a[e] = alo[e]; a[e + 8] = ahi[e]; }
#pragma unroll
    for (int t = 0; t < 8; ++t) {
      const __bf16* pb = Wt + (size_t)(colw + t * 16 + bcol) * D_K + (k0 + koff_b);
      v16bf bb = *(const v16bf*)pb;
      acc[t] = __builtin_amdgcn_wmma_f32_16x16x32_bf16(
          false, a, false, bb, (short)0, acc[t], false, false);
    }
  }

  __shared__ float s_sum[16];
  __shared__ float s_sq[16];
  if (threadIdx.x < 16) { s_sum[threadIdx.x] = 0.f; s_sq[threadIdx.x] = 0.f; }
  __syncthreads();

  const int rbase = hi << 3;
  float vals[8][8];
  float psum[8], psq[8];
#pragma unroll
  for (int r = 0; r < 8; ++r) { psum[r] = 0.f; psq[r] = 0.f; }

#pragma unroll
  for (int t = 0; t < 8; ++t) {
    int col = colw + t * 16 + bcol;
    float bias = bfc[col];
#pragma unroll
    for (int r = 0; r < 8; ++r) {
      float h = fmaxf(acc[t][r] + bias, 0.f);                    // ReLU
      float y = h + x[(size_t)(row0 + rbase + r) * D_IN + col];  // residual
      vals[t][r] = y;
      psum[r] += y;
      psq[r]  += y * y;
    }
  }
  // Reduce across the 16 lanes that share the same 8 rows (xor stays in half)
#pragma unroll
  for (int r = 0; r < 8; ++r) {
#pragma unroll
    for (int off = 8; off > 0; off >>= 1) {
      psum[r] += __shfl_xor(psum[r], off, 32);
      psq[r]  += __shfl_xor(psq[r],  off, 32);
    }
  }
  if (bcol == 0) {
#pragma unroll
    for (int r = 0; r < 8; ++r) {
      atomicAdd(&s_sum[rbase + r], psum[r]);  // ds_add_f32
      atomicAdd(&s_sq[rbase + r],  psq[r]);
    }
  }
  __syncthreads();

#pragma unroll
  for (int t = 0; t < 8; ++t) {
    int col = colw + t * 16 + bcol;
    float g = gamma[col], be = beta[col];
#pragma unroll
    for (int r = 0; r < 8; ++r) {
      int rr = rbase + r;
      float mu  = s_sum[rr] * (1.f / (float)D_IN);
      float var = s_sq[rr]  * (1.f / (float)D_IN) - mu * mu;
      float rs  = rsqrtf(var + 1e-5f);
      out[(size_t)(row0 + rr) * D_IN + col] = (vals[t][r] - mu) * rs * g + be;
    }
  }
}

// ---------------------------------------------------------------------------
// Host orchestration
// ---------------------------------------------------------------------------
extern "C" void kernel_launch(void* const* d_in, const int* in_sizes, int n_in,
                              void* d_out, int out_size, void* d_ws, size_t ws_size,
                              hipStream_t stream) {
  (void)in_sizes; (void)n_in; (void)out_size; (void)ws_size;
  const float* x     = (const float*)d_in[0];
  const float* Wq    = (const float*)d_in[1];
  const float* Wk    = (const float*)d_in[2];
  const float* Wv    = (const float*)d_in[3];
  const float* Wfc   = (const float*)d_in[4];
  const float* bfc   = (const float*)d_in[5];
  const float* gamma = (const float*)d_in[6];
  const float* beta  = (const float*)d_in[7];
  float* out = (float*)d_out;

  char* ws = (char*)d_ws;
  size_t off = 0;
  __bf16* xh   = (__bf16*)(ws + off); off += (size_t)MTOT * D_IN * 2;  // 32 MiB
  __bf16* wqT  = (__bf16*)(ws + off); off += (size_t)D_K * D_IN * 2;   // 4 MiB
  __bf16* wkT  = (__bf16*)(ws + off); off += (size_t)D_K * D_IN * 2;
  __bf16* wvT  = (__bf16*)(ws + off); off += (size_t)D_K * D_IN * 2;
  __bf16* wfcT = (__bf16*)(ws + off); off += (size_t)D_IN * D_K * 2;
  __bf16* qh   = (__bf16*)(ws + off); off += (size_t)MTOT * D_K * 2;   // 64 MiB
  __bf16* kh   = (__bf16*)(ws + off); off += (size_t)MTOT * D_K * 2;
  __bf16* vT   = (__bf16*)(ws + off); off += (size_t)MTOT * D_K * 2;   // [D_K][MTOT]
  __bf16* attx = (__bf16*)(ws + off); off += (size_t)MTOT * D_K * 2;
  float*  sc   = (float*) (ws + off); off += (size_t)NSEQ * NSEQ * 4;  // per-batch scores
  __bf16* attn = (__bf16*)(ws + off); off += (size_t)NSEQ * NSEQ * 2;  // per-batch attn

  dim3 blk256(256), blk128(128);

  // Stage 0: precision lowering + weight transposes (B operands column-major)
  k_cast_bf16<<<dim3((MTOT * D_IN) / 256), blk256, 0, stream>>>(x, xh, MTOT * D_IN);
  k_transpose_bf16<<<dim3(D_K / 32, D_IN / 32), blk256, 0, stream>>>(Wq,  wqT,  D_IN, D_K);
  k_transpose_bf16<<<dim3(D_K / 32, D_IN / 32), blk256, 0, stream>>>(Wk,  wkT,  D_IN, D_K);
  k_transpose_bf16<<<dim3(D_K / 32, D_IN / 32), blk256, 0, stream>>>(Wv,  wvT,  D_IN, D_K);
  k_transpose_bf16<<<dim3(D_IN / 32, D_K / 32), blk256, 0, stream>>>(Wfc, wfcT, D_K,  D_IN);

  // Stage 1: QKV projections. q,k row-major; v stored transposed ([D_K][MTOT])
  // so it directly serves as column-major B for attn @ V.
  k_gemm_wmma<<<dim3(MTOT / 64, D_K / 64), blk128, 0, stream>>>(xh, D_IN, wqT, D_IN, qh, D_K, D_IN, 0);
  k_gemm_wmma<<<dim3(MTOT / 64, D_K / 64), blk128, 0, stream>>>(xh, D_IN, wkT, D_IN, kh, D_K, D_IN, 0);
  k_gemm_wmma<<<dim3(MTOT / 64, D_K / 64), blk128, 0, stream>>>(xh, D_IN, wvT, D_IN, vT, MTOT, D_IN, 1);

  // Stage 2: per-batch attention. Scores buffer (16.7 MB f32) stays L2-resident.
  for (int b = 0; b < BATCH; ++b) {
    const __bf16* qb = qh + (size_t)b * NSEQ * D_K;
    const __bf16* kb = kh + (size_t)b * NSEQ * D_K;  // row-major k == column-major k^T
    k_gemm_wmma<<<dim3(NSEQ / 64, NSEQ / 64), blk128, 0, stream>>>(qb, D_K, kb, D_K, sc, NSEQ, D_K, 2);
    k_softmax_row<<<dim3(NSEQ), blk256, 0, stream>>>(sc, attn);
    k_gemm_wmma<<<dim3(NSEQ / 64, D_K / 64), blk128, 0, stream>>>(
        attn, NSEQ, vT + (size_t)b * NSEQ, MTOT,
        attx + (size_t)b * NSEQ * D_K, D_K, NSEQ, 0);
  }

  // Stage 3: FC + bias + ReLU + residual + LayerNorm (fused)
  k_fc_ln<<<dim3(MTOT / 16), blk256, 0, stream>>>(attx, wfcT, bfc, x, gamma, beta, out);
}
